// MAR_block_missingness_48971217109672
// MI455X (gfx1250) — matrix operand order repaired and checked by reading, not compiled, not run on hardware
//
#include <hip/hip_runtime.h>
#include <hip/hip_bf16.h>
#include <math.h>

// Problem constants (match reference)
constexpr int NRr  = 8192;   // N rows of X
constexpr int NTc  = 8192;   // T cols of X
constexpr int RBn  = 32;
constexpr int CBn  = 32;
constexpr int HIDn = 100;
constexpr int DIOn = 1024;   // RB*CB
constexpr int KPAD = 128;    // hidden dim padded to multiple of 32 for WMMA K

typedef __attribute__((ext_vector_type(16))) _Float16 v16h;
typedef __attribute__((ext_vector_type(8)))  _Float16 v8h;
typedef __attribute__((ext_vector_type(8)))  float    v8f;
typedef __attribute__((ext_vector_type(4)))  float    f32x4;

// ---------------------------------------------------------------------------
// A fragment from a zero-padded 16 x ldk f16 matrix in LDS (rows 1..15 = 0).
// CDNA5 16-bit A 16x32 layout: lanes 0-15 (M=lane) hold K={0..7,16..23};
// lanes 16-31 (M=lane-16) hold K={8..15,24..31}.
// => two unconditional 16-byte LDS loads per lane, no exec masking.
// ---------------------------------------------------------------------------
__device__ inline v16h frag_a_pad(const _Float16* Apad, int ldk, int k0, int lane) {
  const int m  = lane & 15;
  const int kb = (lane & 16) ? 8 : 0;
  const _Float16* p = Apad + m * ldk + k0 + kb;
  const v8h lo = *(const v8h*)(p);        // K = k0+kb .. +7
  const v8h hi = *(const v8h*)(p + 16);   // K = k0+kb+16 .. +23
  v16h a;
#pragma unroll
  for (int i = 0; i < 8; ++i) { a[i] = lo[i]; a[i + 8] = hi[i]; }
  return a;
}

// ---------------------------------------------------------------------------
// B fragment gathered from row-major f32 W[Krows x Ncols], converted to f16.
// Addresses are clamped in-range so loads are unconditional; out-of-range
// elements are zeroed with a register select (no exec branching).
// B 32x16 layout: lanes 0-15 hold K={0..7,16..23} for N=n0+lane; lanes 16-31 K={8..15,24..31}.
// ---------------------------------------------------------------------------
__device__ inline v16h frag_b_gather(const float* __restrict__ W, int Krows, int Ncols,
                                     int k0, int n0, int lane) {
  const int  n   = n0 + (lane & 15);
  const int  nc  = (n < Ncols) ? n : (Ncols - 1);
  const bool nok = (n < Ncols);
  const int  kb  = (lane & 16) ? 8 : 0;
  v16h b;
#pragma unroll
  for (int i = 0; i < 16; ++i) {
    const int  k   = k0 + kb + (i < 8 ? i : i + 8);
    const int  kc  = (k < Krows) ? k : (Krows - 1);
    const float v  = W[(size_t)kc * Ncols + nc];       // always in-bounds
    b[i] = (nok && (k < Krows)) ? (_Float16)v : (_Float16)0.0f;
  }
  return b;
}

// ---------------------------------------------------------------------------
// K0: zero workspace accumulators (atomics accumulate -> re-zero every call).
// ---------------------------------------------------------------------------
__global__ __launch_bounds__(256) void zero_ws_kernel(float* __restrict__ blk_sum,
                                                      int* __restrict__ rcnt,
                                                      int* __restrict__ ccnt) {
  const int i = blockIdx.x * 256 + threadIdx.x;
  if (i < DIOn) blk_sum[i] = 0.0f;
  if (i < RBn)  rcnt[i] = 0;
  if (i < CBn)  ccnt[i] = 0;
}

// ---------------------------------------------------------------------------
// K1: bincounts of the sorted segment ids.
// ---------------------------------------------------------------------------
__global__ __launch_bounds__(256) void count_kernel(const int* __restrict__ row_ids,
                                                    const int* __restrict__ col_ids,
                                                    int* __restrict__ rcnt,
                                                    int* __restrict__ ccnt) {
  const int i = blockIdx.x * 256 + threadIdx.x;
  if (i < NRr) {
    atomicAdd(&rcnt[row_ids[i]], 1);
  } else if (i < NRr + NTc) {
    atomicAdd(&ccnt[col_ids[i - NRr]], 1);
  }
}

// ---------------------------------------------------------------------------
// K2: bandwidth kernel (the roofline stage: 256 MB @ 23.3 TB/s ~ 11 us).
// Each thread owns 4 contiguous columns (128-bit non-temporal coalesced
// loads: 512 contiguous bytes per wave-request) and walks a 256-row chunk.
// row_ids for the chunk are staged in LDS; since they are sorted, flushes to
// the 32x32 block-sum table (f32 global atomics) happen only at row-block
// boundaries (~1-2 per thread).
// ---------------------------------------------------------------------------
__global__ __launch_bounds__(256) void block_reduce_kernel(const float* __restrict__ X,
                                                           const int* __restrict__ row_ids,
                                                           const int* __restrict__ col_ids,
                                                           float* __restrict__ blk_sum) {
  __shared__ int rids[256];
  const int tid = threadIdx.x;
  const int t4  = (blockIdx.x * 256 + tid) * 4;   // first of 4 owned columns
  const int r0  = blockIdx.y * 256;               // row chunk start

  rids[tid] = row_ids[r0 + tid];

  const int cb0 = col_ids[t4 + 0];
  const int cb1 = col_ids[t4 + 1];
  const int cb2 = col_ids[t4 + 2];
  const int cb3 = col_ids[t4 + 3];

  __syncthreads();

  float ax = 0.0f, ay = 0.0f, az = 0.0f, aw = 0.0f;
  int cur = rids[0];

#pragma unroll 8
  for (int r = 0; r < 256; ++r) {
    const int rb = rids[r];
    if (rb != cur) {
      atomicAdd(&blk_sum[cur * CBn + cb0], ax);
      atomicAdd(&blk_sum[cur * CBn + cb1], ay);
      atomicAdd(&blk_sum[cur * CBn + cb2], az);
      atomicAdd(&blk_sum[cur * CBn + cb3], aw);
      ax = ay = az = aw = 0.0f;
      cur = rb;
    }
    const f32x4 v = __builtin_nontemporal_load(
        (const f32x4*)(X + (size_t)(r0 + r) * NTc + t4));   // global_load_b128 (NT)
    ax += v.x; ay += v.y; az += v.z; aw += v.w;
  }
  atomicAdd(&blk_sum[cur * CBn + cb0], ax);
  atomicAdd(&blk_sum[cur * CBn + cb1], ay);
  atomicAdd(&blk_sum[cur * CBn + cb2], az);
  atomicAdd(&blk_sum[cur * CBn + cb3], aw);
}

// ---------------------------------------------------------------------------
// K3: single-workgroup MLP on the 32x32 block means using WMMA f16->f32,
// plus the two cumsum outputs.
//   out[0..1023]     propensity (sigmoid)
//   out[1024..1056]  row_cumsum (33, as float)
//   out[1057..1089]  col_cumsum (33, as float)
// ---------------------------------------------------------------------------
__global__ __launch_bounds__(256) void mar_mlp_kernel(const float* __restrict__ blk_sum,
                                                      const int* __restrict__ rcnt,
                                                      const int* __restrict__ ccnt,
                                                      const float* __restrict__ W1,
                                                      const float* __restrict__ b1,
                                                      const float* __restrict__ W2,
                                                      const float* __restrict__ b2,
                                                      const float* __restrict__ W3,
                                                      const float* __restrict__ b3,
                                                      float* __restrict__ out) {
  // Zero-padded A matrices (row 0 = activations, rows 1..15 = 0)
  __shared__ __align__(16) _Float16 h0pad[16 * DIOn];   // 32 KB
  __shared__ __align__(16) _Float16 h1pad[16 * KPAD];   // 4 KB
  __shared__ __align__(16) _Float16 h2pad[16 * KPAD];   // 4 KB

  const int tid  = threadIdx.x;
  const int wave = tid >> 5;
  const int lane = tid & 31;

  // zero all pads (as dwords), then fill row 0 of h0pad
  {
    int* z0 = (int*)h0pad;
    int* z1 = (int*)h1pad;
    int* z2 = (int*)h2pad;
#pragma unroll
    for (int i = tid; i < 16 * DIOn / 2; i += 256) z0[i] = 0;
    if (tid < 16 * KPAD / 2) { z1[tid] = 0; z2[tid] = 0; }
  }
  __syncthreads();

#pragma unroll
  for (int i = tid; i < DIOn; i += 256) {
    const int r = i >> 5, c = i & 31;
    float cnt = (float)rcnt[r] * (float)ccnt[c];
    cnt = fmaxf(cnt, 1.0f);
    h0pad[i] = (_Float16)(blk_sum[i] / cnt);
  }
  __syncthreads();

  // ---- layer 1: (1x1024) @ W1(1024x100), relu.  7 N-tiles, K = 32 x 32 ----
  if (wave < 7) {
    const int n0 = wave * 16;
    v8f c = {};
#pragma unroll 2
    for (int k0 = 0; k0 < DIOn; k0 += 32) {
      const v16h a = frag_a_pad(h0pad, DIOn, k0, lane);
      const v16h b = frag_b_gather(W1, DIOn, HIDn, k0, n0, lane);
      c = __builtin_amdgcn_wmma_f32_16x16x32_f16(false, a, false, b, (short)0, c, false, false);
    }
    if (lane < 16) {                 // M=0 row lives in c[0] of lanes 0..15
      const int n = n0 + lane;
      if (n < HIDn) h1pad[n] = (_Float16)fmaxf(c[0] + b1[n], 0.0f);
    }
  }
  __syncthreads();

  // ---- layer 2: (1x100) @ W2(100x100), relu.  K padded to 128 ----
  if (wave < 7) {
    const int n0 = wave * 16;
    v8f c = {};
#pragma unroll 2
    for (int k0 = 0; k0 < KPAD; k0 += 32) {
      const v16h a = frag_a_pad(h1pad, KPAD, k0, lane);
      const v16h b = frag_b_gather(W2, HIDn, HIDn, k0, n0, lane);
      c = __builtin_amdgcn_wmma_f32_16x16x32_f16(false, a, false, b, (short)0, c, false, false);
    }
    if (lane < 16) {
      const int n = n0 + lane;
      if (n < HIDn) h2pad[n] = (_Float16)fmaxf(c[0] + b2[n], 0.0f);
    }
  }
  __syncthreads();

  // ---- layer 3: (1x100) @ W3(100x1024), sigmoid.  64 N-tiles over 8 waves --
  for (int tile = wave; tile < 64; tile += 8) {
    const int n0 = tile * 16;
    v8f c = {};
#pragma unroll 2
    for (int k0 = 0; k0 < KPAD; k0 += 32) {
      const v16h a = frag_a_pad(h2pad, KPAD, k0, lane);
      const v16h b = frag_b_gather(W3, HIDn, DIOn, k0, n0, lane);
      c = __builtin_amdgcn_wmma_f32_16x16x32_f16(false, a, false, b, (short)0, c, false, false);
    }
    if (lane < 16) {
      const int n = n0 + lane;
      const float v = c[0] + b3[n];
      out[n] = 1.0f / (1.0f + __expf(-v));
    }
  }

  // ---- cumsums (33 entries each), serial on one thread: trivial work ----
  if (tid == 0) {
    int run = 0;
    out[DIOn] = 0.0f;
    for (int i = 0; i < RBn; ++i) { run += rcnt[i]; out[DIOn + 1 + i] = (float)run; }
    run = 0;
    out[DIOn + 33] = 0.0f;
    for (int i = 0; i < CBn; ++i) { run += ccnt[i]; out[DIOn + 34 + i] = (float)run; }
  }
}

// ---------------------------------------------------------------------------
extern "C" void kernel_launch(void* const* d_in, const int* in_sizes, int n_in,
                              void* d_out, int out_size, void* d_ws, size_t ws_size,
                              hipStream_t stream) {
  const float* X       = (const float*)d_in[0];
  const int*   row_ids = (const int*)  d_in[1];
  const int*   col_ids = (const int*)  d_in[2];
  const float* W1      = (const float*)d_in[3];
  const float* b1      = (const float*)d_in[4];
  const float* W2      = (const float*)d_in[5];
  const float* b2      = (const float*)d_in[6];
  const float* W3      = (const float*)d_in[7];
  const float* b3      = (const float*)d_in[8];
  float* out = (float*)d_out;

  // workspace: blk_sum[1024] f32, then rcnt[32], ccnt[32] i32  (~4.4 KB)
  float* blk_sum = (float*)d_ws;
  int*   rcnt    = (int*)((char*)d_ws + DIOn * sizeof(float));
  int*   ccnt    = rcnt + RBn;

  zero_ws_kernel<<<dim3(4), dim3(256), 0, stream>>>(blk_sum, rcnt, ccnt);
  count_kernel<<<dim3((NRr + NTc) / 256), dim3(256), 0, stream>>>(row_ids, col_ids, rcnt, ccnt);
  block_reduce_kernel<<<dim3(NTc / 1024, NRr / 256), dim3(256), 0, stream>>>(X, row_ids, col_ids, blk_sum);
  mar_mlp_kernel<<<dim3(1), dim3(256), 0, stream>>>(blk_sum, rcnt, ccnt,
                                                    W1, b1, W2, b2, W3, b3, out);
}
